// HybridSoftmaxLinearAttention_56530359549997
// MI455X (gfx1250) — matrix operand order repaired and checked by reading, not compiled, not run on hardware
//
#include <hip/hip_runtime.h>
#include <hip/hip_bf16.h>
#include <math.h>

// ---------------- problem constants ----------------
#define BDIM 2
#define HQ   32
#define HKV  8
#define GQ   4
#define TSEQ 4096
#define DDIM 64
#define FDIM 64
#define WWIN 64
#define NBLK 64

// padded LDS leading dims (halfs). 72 halfs = 36 dwords, 136 halfs = 68 dwords:
// both 16B-aligned row strides whose dword stride mod 64 spreads 16 lanes over
// all banks -> conflict-free ds_load_b128 fragment reads.
#define LD64  72
#define LD128 136

typedef __attribute__((ext_vector_type(16))) __bf16 v16bf;
typedef __attribute__((ext_vector_type(8)))  float  v8f;

__device__ __forceinline__ __bf16 bf(float x) { return (__bf16)x; }

__device__ __forceinline__ v8f wmma_bf16(v16bf a, v16bf b, v8f c) {
  // D = A(16x32 bf16) x B(32x16 bf16) + C(f32)
  return __builtin_amdgcn_wmma_f32_16x16x32_bf16(false, a, false, b, (short)0, c,
                                                 false, false);
}

// ---- fragment gathers from row-major LDS tiles (wave32 layouts, ISA 7.12.2) ----
// A 16x32 bf16: lane L holds row M=L%16 (lanes>=16: K offset +8); per lane the
// 16 halfs form two contiguous 16B runs -> 2x ds_load_b128.
__device__ __forceinline__ v16bf frag_A(const __bf16* p, int m0, int k0, int ld) {
  const int lane = threadIdx.x & 31;
  const int m = m0 + (lane & 15);
  const int kb = (lane < 16) ? 0 : 8;
  v16bf r;
#pragma unroll
  for (int h = 0; h < 16; ++h) {
    const int j = h >> 1;
    const int kk = ((j >= 4) ? 16 : 0) + kb + 2 * (j & 3) + (h & 1);
    r[h] = p[m * ld + k0 + kk];
  }
  return r;
}
// B fragment with source stored transposed: logical B[k][n] = src[n][k].
// Per lane: n fixed (lanes>=16: K offset +8); two contiguous 16B runs -> 2x ds_load_b128.
__device__ __forceinline__ v16bf frag_BT(const __bf16* p, int k0, int n0, int ld) {
  const int lane = threadIdx.x & 31;
  const int n = n0 + (lane & 15);
  const int kb = (lane < 16) ? 0 : 8;
  v16bf r;
#pragma unroll
  for (int h = 0; h < 16; ++h) r[h] = p[n * ld + k0 + ((h >= 8) ? 16 : 0) + kb + (h & 7)];
  return r;
}

// =====================================================================
// Kernel 1: per (b, hkv, block) -> S_blk = phi(k)^T V  [F x D], z_blk [F]
// =====================================================================
__global__ void __launch_bounds__(256) k1_kv_state(
    const float* __restrict__ kg, const float* __restrict__ vg,
    const float* __restrict__ projg, float* __restrict__ S_blk,
    float* __restrict__ z_blk) {
  __shared__ __bf16 ls_k[WWIN * LD64];    // scaled k, row-major [w][d]
  __shared__ __bf16 ls_vt[DDIM * LD64];   // v transposed   [d][w]
  __shared__ __bf16 ls_p[FDIM * LD64];    // proj row-major [f][d]
  __shared__ __bf16 ls_pkt[FDIM * LD64];  // phi(k) transposed [f][w]
  __shared__ float  ls_sqn[WWIN];
  __shared__ float  ls_z[FDIM];

  const int t = threadIdx.x;
  const int wg = blockIdx.x;  // = (b*HKV + h)*NBLK + c
  const int c = wg % NBLK;
  const int bh = wg / NBLK;
  const float s1 = 0.3535533906f;  // 64^-0.25

  if (t < WWIN) { ls_sqn[t] = 0.f; ls_z[t] = 0.f; }
  __syncthreads();

  {
    const int row = t >> 2, seg = (t & 3) * 16;
    const float* kp = kg + ((size_t)bh * TSEQ + (size_t)c * WWIN + row) * DDIM + seg;
    const float* vp = vg + ((size_t)bh * TSEQ + (size_t)c * WWIN + row) * DDIM + seg;
    float ss = 0.f;
#pragma unroll
    for (int i = 0; i < 16; ++i) {
      const float kv = kp[i] * s1;
      ss += kv * kv;
      ls_k[row * LD64 + seg + i] = bf(kv);
      ls_vt[(seg + i) * LD64 + row] = bf(vp[i]);  // transpose on fill
    }
    atomicAdd(&ls_sqn[row], 0.5f * ss);
    const int f0 = t >> 2;  // proj row, 16 contiguous d per thread
#pragma unroll
    for (int i = 0; i < 16; ++i)
      ls_p[f0 * LD64 + seg + i] = bf(projg[f0 * DDIM + seg + i]);
  }
  __syncthreads();

  const int wave = t >> 5, lane = t & 31;
  const int hi8 = (lane < 16) ? 0 : 8;

  // phi(k) = exp(clip(ks @ proj^T - sqn)) / sqrt(F); stored transposed [f][w]
#pragma unroll
  for (int i = 0; i < 2; ++i) {
    const int idx = wave * 2 + i;
    const int m0 = (idx >> 2) * 16, n0 = (idx & 3) * 16;
    v8f acc = {};
    acc = wmma_bf16(frag_A(ls_k, m0, 0, LD64),  frag_BT(ls_p, 0, n0, LD64),  acc);
    acc = wmma_bf16(frag_A(ls_k, m0, 32, LD64), frag_BT(ls_p, 32, n0, LD64), acc);
#pragma unroll
    for (int r = 0; r < 8; ++r) {
      const int M = m0 + r + hi8, N = n0 + (lane & 15);  // M = w, N = f
      float lg = fminf(fmaxf(acc[r] - ls_sqn[M], -20.f), 20.f);
      const float pv = __expf(lg) * 0.125f;  // 1/sqrt(64)
      ls_pkt[N * LD64 + M] = bf(pv);
      atomicAdd(&ls_z[N], pv);
    }
  }
  __syncthreads();

  // S_blk[f][d] = sum_w pk[w][f] * v[w][d]  (A = pkt row-major, B = vt transposed)
  float* Sout = S_blk + (size_t)wg * (FDIM * DDIM);
#pragma unroll
  for (int i = 0; i < 2; ++i) {
    const int idx = wave * 2 + i;
    const int m0 = (idx >> 2) * 16, n0 = (idx & 3) * 16;
    v8f acc = {};
    acc = wmma_bf16(frag_A(ls_pkt, m0, 0, LD64),  frag_BT(ls_vt, 0, n0, LD64),  acc);
    acc = wmma_bf16(frag_A(ls_pkt, m0, 32, LD64), frag_BT(ls_vt, 32, n0, LD64), acc);
#pragma unroll
    for (int r = 0; r < 8; ++r) {
      const int M = m0 + r + hi8, N = n0 + (lane & 15);
      Sout[(size_t)M * DDIM + N] = acc[r];
    }
  }
  if (t < FDIM) z_blk[(size_t)wg * FDIM + t] = ls_z[t];
}

// =====================================================================
// Kernel 2: exclusive-by-2 prefix sum: S_used[c] = sum_{j<=c-2} S_blk[j]
// =====================================================================
__global__ void __launch_bounds__(256) k2_scan(
    const float* __restrict__ S_blk, const float* __restrict__ z_blk,
    float* __restrict__ S_used, float* __restrict__ z_used) {
  const int t = threadIdx.x;
  const int bh = blockIdx.x;
  const int e0 = t * 16;  // 4096 elements / 256 threads
  float acc[16];
#pragma unroll
  for (int i = 0; i < 16; ++i) acc[i] = 0.f;
  float az = 0.f;
  for (int c = 0; c < NBLK; ++c) {
    const size_t bw = ((size_t)bh * NBLK + c) * (FDIM * DDIM);
#pragma unroll
    for (int i = 0; i < 16; ++i) S_used[bw + e0 + i] = acc[i];
    if (t < FDIM) z_used[((size_t)bh * NBLK + c) * FDIM + t] = az;
    if (c >= 1) {
      const size_t br = ((size_t)bh * NBLK + (c - 1)) * (FDIM * DDIM);
#pragma unroll
      for (int i = 0; i < 16; ++i) acc[i] += S_blk[br + e0 + i];
      if (t < FDIM) az += z_blk[((size_t)bh * NBLK + (c - 1)) * FDIM + t];
    }
  }
}

// =====================================================================
// Kernel 3: per (b, hkv, g, block) -> output 64x64 tile
// =====================================================================
__global__ void __launch_bounds__(256) k3_out(
    const float* __restrict__ qg, const float* __restrict__ kg,
    const float* __restrict__ vg, const float* __restrict__ projg,
    const float* __restrict__ S_used, const float* __restrict__ z_used,
    float* __restrict__ outg) {
  __shared__ __bf16 ls_q[WWIN * LD64];        // scaled q row-major [w][d]
  __shared__ __bf16 ls_k[2 * WWIN * LD64];    // [prev|cur] raw k  [x][d]
  __shared__ __bf16 ls_vt[DDIM * LD128];      // [prev|cur] raw v transposed [d][x]
  __shared__ __bf16 ls_p[FDIM * LD64];        // proj [f][d]
  __shared__ __bf16 ls_pq[WWIN * LD64];       // phi(q) [w][f]
  __shared__ __bf16 ls_a[WWIN * LD128];       // exp(scores) [w][x]
  __shared__ __bf16 ls_St[DDIM * LD64];       // S_used transposed [d][f]
  __shared__ float  ls_sqn[WWIN], ls_z[FDIM], ls_rs[WWIN], ls_den[WWIN];

  const int t = threadIdx.x;
  const int wg = blockIdx.x;  // ((b*HKV+h)*GQ + g)*NBLK + c
  const int c = wg % NBLK;
  const int g = (wg / NBLK) % GQ;
  const int h = (wg / (NBLK * GQ)) % HKV;
  const int b = wg / (NBLK * GQ * HKV);
  const int bh = b * HKV + h;
  const int qh = h * GQ + g;
  const float s1 = 0.3535533906f;  // 64^-0.25 == 64^0.25/sqrt(64)

  if (t < WWIN) { ls_sqn[t] = 0.f; ls_rs[t] = 0.f; }
  __syncthreads();

  {
    // q block (scaled) + sq_norm
    const int row = t >> 2, seg = (t & 3) * 16;
    const float* qp =
        qg + (((size_t)b * HQ + qh) * TSEQ + (size_t)c * WWIN + row) * DDIM + seg;
    float ss = 0.f;
#pragma unroll
    for (int i = 0; i < 16; ++i) {
      const float qv = qp[i] * s1;
      ss += qv * qv;
      ls_q[row * LD64 + seg + i] = bf(qv);
    }
    atomicAdd(&ls_sqn[row], 0.5f * ss);

    // k/v [prev|cur]: 128 rows x 64, 32 elems per thread; v transposed on fill
    const int row2 = t >> 1, seg2 = (t & 1) * 32;
    const size_t kvb = (size_t)bh * TSEQ * DDIM;
    if (row2 < WWIN) {
      if (c == 0) {
#pragma unroll
        for (int i = 0; i < 32; ++i) {
          ls_k[row2 * LD64 + seg2 + i] = bf(0.f);
          ls_vt[(seg2 + i) * LD128 + row2] = bf(0.f);
        }
      } else {
        const float* kp = kg + kvb + ((size_t)(c - 1) * WWIN + row2) * DDIM + seg2;
        const float* vp = vg + kvb + ((size_t)(c - 1) * WWIN + row2) * DDIM + seg2;
#pragma unroll
        for (int i = 0; i < 32; ++i) {
          ls_k[row2 * LD64 + seg2 + i] = bf(kp[i]);
          ls_vt[(seg2 + i) * LD128 + row2] = bf(vp[i]);
        }
      }
    } else {
      const int rr = row2 - WWIN;
      const float* kp = kg + kvb + ((size_t)c * WWIN + rr) * DDIM + seg2;
      const float* vp = vg + kvb + ((size_t)c * WWIN + rr) * DDIM + seg2;
#pragma unroll
      for (int i = 0; i < 32; ++i) {
        ls_k[row2 * LD64 + seg2 + i] = bf(kp[i]);
        ls_vt[(seg2 + i) * LD128 + row2] = bf(vp[i]);
      }
    }

    // proj (row-major) + S_used (transposed on fill)
    const int f0 = t >> 2;              // row, 16 contiguous cols per thread
    const int d0 = (t & 3) * 16;
    const float* Sp = S_used + ((size_t)bh * NBLK + c) * (FDIM * DDIM);
#pragma unroll
    for (int i = 0; i < 16; ++i) {
      ls_p[f0 * LD64 + d0 + i] = bf(projg[f0 * DDIM + d0 + i]);
      ls_St[(d0 + i) * LD64 + f0] = bf(Sp[f0 * DDIM + d0 + i]);
    }
    if (t < FDIM) ls_z[t] = z_used[((size_t)bh * NBLK + c) * FDIM + t];
  }
  __syncthreads();

  const int wave = t >> 5, lane = t & 31;
  const int hi8 = (lane < 16) ? 0 : 8;

  // G1: pq = phi(q)
#pragma unroll
  for (int i = 0; i < 2; ++i) {
    const int idx = wave * 2 + i;
    const int m0 = (idx >> 2) * 16, n0 = (idx & 3) * 16;
    v8f acc = {};
    acc = wmma_bf16(frag_A(ls_q, m0, 0, LD64),  frag_BT(ls_p, 0, n0, LD64),  acc);
    acc = wmma_bf16(frag_A(ls_q, m0, 32, LD64), frag_BT(ls_p, 32, n0, LD64), acc);
#pragma unroll
    for (int r = 0; r < 8; ++r) {
      const int M = m0 + r + hi8, N = n0 + (lane & 15);
      float lg = fminf(fmaxf(acc[r] - ls_sqn[M], -20.f), 20.f);
      ls_pq[M * LD64 + N] = bf(__expf(lg) * 0.125f);
    }
  }
  __syncthreads();

  // den_lin = pq . z_used  (waves 0-1, overlapped with G2 on other waves)
  if (t < WWIN) {
    float s = 0.f;
#pragma unroll 8
    for (int f = 0; f < FDIM; ++f) s += (float)ls_pq[t * LD64 + f] * ls_z[f];
    ls_den[t] = s;
  }

  // G2: scores = qs @ kk2^T (*s1) -> masked exp -> a, rowsums
#pragma unroll
  for (int i = 0; i < 4; ++i) {
    const int idx = wave * 4 + i;
    const int m0 = (idx >> 3) * 16, x0 = (idx & 7) * 16;
    v8f acc = {};
    acc = wmma_bf16(frag_A(ls_q, m0, 0, LD64),  frag_BT(ls_k, 0, x0, LD64),  acc);
    acc = wmma_bf16(frag_A(ls_q, m0, 32, LD64), frag_BT(ls_k, 32, x0, LD64), acc);
#pragma unroll
    for (int r = 0; r < 8; ++r) {
      const int M = m0 + r + hi8, X = x0 + (lane & 15);
      const float s = acc[r] * s1;
      const bool ok = (X <= WWIN + M) && ((c > 0) || (X >= WWIN));
      float av = 0.f;
      if (ok) av = __expf(fminf(fmaxf(s, -30.f), 30.f));
      ls_a[M * LD128 + X] = bf(av);
      if (ok) atomicAdd(&ls_rs[M], av);
    }
  }
  __syncthreads();

  // G3 + G4 fused: num = pq @ S_used + a @ vv2 ; out = num / den
  float* op = outg + (((size_t)b * HQ + qh) * TSEQ + (size_t)c * WWIN) * DDIM;
#pragma unroll
  for (int i = 0; i < 2; ++i) {
    const int idx = wave * 2 + i;
    const int m0 = (idx >> 2) * 16, n0 = (idx & 3) * 16;
    v8f acc = {};
    acc = wmma_bf16(frag_A(ls_pq, m0, 0, LD64),  frag_BT(ls_St, 0, n0, LD64),  acc);
    acc = wmma_bf16(frag_A(ls_pq, m0, 32, LD64), frag_BT(ls_St, 32, n0, LD64), acc);
#pragma unroll
    for (int k0 = 0; k0 < 2 * WWIN; k0 += 32)
      acc = wmma_bf16(frag_A(ls_a, m0, k0, LD128), frag_BT(ls_vt, k0, n0, LD128), acc);
#pragma unroll
    for (int r = 0; r < 8; ++r) {
      const int M = m0 + r + hi8, N = n0 + (lane & 15);
      const float den = ls_rs[M] + ls_den[M] + 1e-6f;
      op[(size_t)M * DDIM + N] = acc[r] / den;
    }
  }
}

// =====================================================================
extern "C" void kernel_launch(void* const* d_in, const int* in_sizes, int n_in,
                              void* d_out, int out_size, void* d_ws, size_t ws_size,
                              hipStream_t stream) {
  (void)in_sizes; (void)n_in; (void)out_size; (void)ws_size;
  const float* q = (const float*)d_in[0];
  const float* kk = (const float*)d_in[1];
  const float* vv = (const float*)d_in[2];
  const float* proj = (const float*)d_in[3];
  float* out = (float*)d_out;

  float* ws = (float*)d_ws;
  const size_t SSZ = (size_t)BDIM * HKV * NBLK * FDIM * DDIM;  // 4,194,304 f32
  const size_t ZSZ = (size_t)BDIM * HKV * NBLK * FDIM;         // 65,536 f32
  float* S_blk  = ws;
  float* S_used = ws + SSZ;
  float* z_blk  = ws + 2 * SSZ;
  float* z_used = z_blk + ZSZ;

  k1_kv_state<<<BDIM * HKV * NBLK, 256, 0, stream>>>(kk, vv, proj, S_blk, z_blk);
  k2_scan<<<BDIM * HKV, 256, 0, stream>>>(S_blk, z_blk, S_used, z_used);
  k3_out<<<BDIM * HKV * GQ * NBLK, 256, 0, stream>>>(q, kk, vv, proj, S_used,
                                                     z_used, out);
}
